// Attention_22041772163436
// MI455X (gfx1250) — matrix operand order repaired and checked by reading, not compiled, not run on hardware
//
#include <hip/hip_runtime.h>
#include <hip/hip_bf16.h>
#include <math.h>

// ---------------------------------------------------------------------------
// Bahdanau-style attention, fused for MI455X (gfx1250, wave32, WMMA).
//   score  = tanh(X @ W + Wb)          : [B,T,256] -- never materialized
//   logits = score @ Vw + Vb           : [B,T]
//   attn   = softmax_T(logits)
//   out    = sum_t attn * X            : [B,256]
// ---------------------------------------------------------------------------

typedef __attribute__((ext_vector_type(16))) __bf16 v16bf;
typedef __attribute__((ext_vector_type(8)))  float  v8f;

#define BATCH  64
#define TLEN   4096
#define DIM    256           // D == UNITS
#define BT     (BATCH * TLEN)
#define ROWS_PER_WG 128      // 8 waves x one 16-row M-tile each
#define NT     16            // N tiles of 16 (covers 256 units)
#define KS     8             // K steps of 32 (covers 256 dims)
#define TCHUNK 8             // T-chunks for the weighted-sum pass
#define NPART  (TCHUNK * 4)  // partial-context chunks (chunk x row-phase)

// Branch-free tanh: tanh(x) = sign(x) * (1 - 2/(exp(2|x|)+1)).
// exp -> TRANS pipe (co-executes with VALU/WMMA); rcp -> v_rcp_f32.
// Saturates cleanly for large |x| (exp->inf => 1), exact at 0.
__device__ __forceinline__ float fast_tanh(float x) {
  float ax = __builtin_fabsf(x);
  float e  = __expf(2.0f * ax);
  float r  = 1.0f - 2.0f * __builtin_amdgcn_rcpf(e + 1.0f);
  return __builtin_copysignf(r, x);
}

// --------------------------------------------------------------------------
// Kernel 0: convert W (fp32, [K=256][N=256] row-major) to bf16, pre-swizzled
// into WMMA B-fragment order: frag(nt,ks) -> 32 lanes x 16 halves contiguous.
// B layout (16-bit, 32x16): lane L holds column N = L&15; K = 8*(L>=16) + 16*(e>>3) + (e&7).
// --------------------------------------------------------------------------
__global__ __launch_bounds__(256) void prep_w_bf16(const float* __restrict__ W,
                                                   __bf16* __restrict__ Wsw) {
  int idx  = blockIdx.x * 256 + threadIdx.x;   // 0 .. 65535
  int e    = idx & 15;
  int lane = (idx >> 4) & 31;
  int frag = idx >> 9;                         // nt*KS + ks
  int ks   = frag & (KS - 1);
  int nt   = frag >> 3;
  int k    = ks * 32 + ((lane >> 4) << 3) + ((e >> 3) << 4) + (e & 7);
  int n    = nt * 16 + (lane & 15);
  Wsw[idx] = (__bf16)W[k * DIM + n];
}

// --------------------------------------------------------------------------
// Kernel 1: fused  logits[r] = Vw . tanh(X[r,:] @ W + Wb) + Vb
// Grid: BT/ROWS_PER_WG workgroups of 256 threads (8 waves).
// Each wave owns a 16-row M-tile; A panel lives in VGPRs; W lives in LDS.
// --------------------------------------------------------------------------
__global__ __launch_bounds__(256) void score_logits(
    const float* __restrict__ X, const __bf16* __restrict__ Wsw,
    const float* __restrict__ Wb, const float* __restrict__ Vw,
    const float* __restrict__ Vb, float* __restrict__ logits)
{
  __shared__ __bf16 Wl[NT * KS * 32 * 16];     // 65536 halves = 128 KB

  const int tid = threadIdx.x;

  // ---- stage swizzled bf16 W into LDS (coalesced b128 copies) ----
  {
    const uint4* src = reinterpret_cast<const uint4*>(Wsw);
    uint4*       dst = reinterpret_cast<uint4*>(Wl);
    #pragma unroll 4
    for (int i = tid; i < (NT * KS * 32 * 16) / 8; i += 256) dst[i] = src[i];
  }

  const int wave  = tid >> 5;    // 0..7 == M-tile index
  const int lane  = tid & 31;
  const int lhalf = lane >> 4;   // 0 | 1
  const int lmod  = lane & 15;

  // ---- load this lane's A-panel (one X row, bf16) into VGPRs ----
  // A layout (16-bit, 16x32): M = lane&15; K = 16*(e>>3) + 8*(lane>=16) + (e&7)
  const float* xrow =
      X + ((long)blockIdx.x * ROWS_PER_WG + wave * 16 + lmod) * (long)DIM;
  v16bf afrag[KS];
  #pragma unroll
  for (int ks = 0; ks < KS; ++ks) {
    const int kb = ks * 32 + lhalf * 8;
    float4 f0 = *reinterpret_cast<const float4*>(xrow + kb);
    float4 f1 = *reinterpret_cast<const float4*>(xrow + kb + 4);
    float4 f2 = *reinterpret_cast<const float4*>(xrow + kb + 16);
    float4 f3 = *reinterpret_cast<const float4*>(xrow + kb + 20);
    v16bf a;
    a[0]  = (__bf16)f0.x; a[1]  = (__bf16)f0.y; a[2]  = (__bf16)f0.z; a[3]  = (__bf16)f0.w;
    a[4]  = (__bf16)f1.x; a[5]  = (__bf16)f1.y; a[6]  = (__bf16)f1.z; a[7]  = (__bf16)f1.w;
    a[8]  = (__bf16)f2.x; a[9]  = (__bf16)f2.y; a[10] = (__bf16)f2.z; a[11] = (__bf16)f2.w;
    a[12] = (__bf16)f3.x; a[13] = (__bf16)f3.y; a[14] = (__bf16)f3.z; a[15] = (__bf16)f3.w;
    afrag[ks] = a;
  }

  __syncthreads();

  // ---- GEMM + tanh + V-dot, one 16x16 C tile at a time ----
  float plog[8] = {};
  #pragma unroll 1
  for (int nt = 0; nt < NT; ++nt) {
    v8f acc = {0.f, 0.f, 0.f, 0.f, 0.f, 0.f, 0.f, 0.f};
    #pragma unroll
    for (int ks = 0; ks < KS; ++ks) {
      const __bf16* bp = Wl + (((nt * KS + ks) * 32 + lane) << 4);
      union { v16bf v; uint4 q[2]; } b;
      b.q[0] = *reinterpret_cast<const uint4*>(bp);       // 32B aligned
      b.q[1] = *reinterpret_cast<const uint4*>(bp + 8);
      acc = __builtin_amdgcn_wmma_f32_16x16x32_bf16(
          false, afrag[ks], false, b.v, (short)0, acc, false, false);
    }
    // C layout: acc[i] -> (M = i + 8*lhalf, N = lmod) within this tile
    const int   n  = nt * 16 + lmod;
    const float wb = Wb[n];
    const float vw = Vw[n];
    #pragma unroll
    for (int i = 0; i < 8; ++i) plog[i] += fast_tanh(acc[i] + wb) * vw;
  }

  // ---- per-row logit: butterfly-sum over the 16 lanes of each half ----
  #pragma unroll
  for (int i = 0; i < 8; ++i) {
    #pragma unroll
    for (int off = 1; off < 16; off <<= 1)
      plog[i] += __shfl_xor(plog[i], off, 32);
  }
  if (lmod == 0) {
    const float vb  = Vb[0];
    const long base = (long)blockIdx.x * ROWS_PER_WG + wave * 16 + lhalf * 8;
    #pragma unroll
    for (int i = 0; i < 8; ++i) logits[base + i] = plog[i] + vb;
  }
}

// --------------------------------------------------------------------------
// Kernel 2: softmax over T per batch. One block per batch; deterministic
// fixed-order tree reductions in LDS; logits kept in registers (16/thread).
// --------------------------------------------------------------------------
__global__ __launch_bounds__(256) void softmax_t(const float* __restrict__ logits,
                                                 float* __restrict__ attn) {
  __shared__ float red[256];
  const int b = blockIdx.x, tid = threadIdx.x;
  float v[TLEN / 256];
  float mx = -INFINITY;
  #pragma unroll
  for (int i = 0; i < TLEN / 256; ++i) {
    v[i] = logits[(long)b * TLEN + i * 256 + tid];
    mx = fmaxf(mx, v[i]);
  }
  red[tid] = mx; __syncthreads();
  for (int s = 128; s > 0; s >>= 1) {
    if (tid < s) red[tid] = fmaxf(red[tid], red[tid + s]);
    __syncthreads();
  }
  mx = red[0]; __syncthreads();

  float sum = 0.f;
  #pragma unroll
  for (int i = 0; i < TLEN / 256; ++i) { v[i] = expf(v[i] - mx); sum += v[i]; }
  red[tid] = sum; __syncthreads();
  for (int s = 128; s > 0; s >>= 1) {
    if (tid < s) red[tid] += red[tid + s];
    __syncthreads();
  }
  const float inv = 1.f / red[0];
  #pragma unroll
  for (int i = 0; i < TLEN / 256; ++i)
    attn[(long)b * TLEN + i * 256 + tid] = v[i] * inv;
}

// --------------------------------------------------------------------------
// Kernel 3: partial context over a T-chunk, b128 global loads.
// block = (b, chunk); thread = (row-phase r 0..3, float4-column q 0..63).
// Each thread streams rows t = r, r+4, ... of its chunk with float4 loads;
// per-(chunk,phase) partials are combined later in fixed order.
// --------------------------------------------------------------------------
__global__ __launch_bounds__(256) void weighted_sum(const float* __restrict__ X,
                                                    const float* __restrict__ attn,
                                                    float* __restrict__ partials) {
  __shared__ float a_s[TLEN / TCHUNK];         // 512 floats
  const int b = blockIdx.x >> 3, c = blockIdx.x & (TCHUNK - 1);
  const int tid = threadIdx.x;
  const int t0 = c * (TLEN / TCHUNK);
  for (int i = tid; i < TLEN / TCHUNK; i += 256)
    a_s[i] = attn[(long)b * TLEN + t0 + i];
  __syncthreads();

  const int q = tid & 63;          // float4 column group: d = 4q .. 4q+3
  const int r = tid >> 6;          // row phase 0..3
  const float4* xp =
      reinterpret_cast<const float4*>(X + ((long)b * TLEN + t0) * DIM) + q;
  float4 acc = {0.f, 0.f, 0.f, 0.f};
  #pragma unroll 4
  for (int t = r; t < TLEN / TCHUNK; t += 4) {
    float4 x = xp[(long)t * (DIM / 4)];
    float  a = a_s[t];
    acc.x = fmaf(a, x.x, acc.x);
    acc.y = fmaf(a, x.y, acc.y);
    acc.z = fmaf(a, x.z, acc.z);
    acc.w = fmaf(a, x.w, acc.w);
  }
  const int p = c * 4 + r;         // partial chunk id 0..31
  float4* pout = reinterpret_cast<float4*>(
      partials + ((long)p * BATCH + b) * DIM) + q;
  *pout = acc;
}

// --------------------------------------------------------------------------
// Kernel 4: deterministic final reduce of the NPART partials -> d_out.
// --------------------------------------------------------------------------
__global__ __launch_bounds__(256) void reduce_partials(const float* __restrict__ partials,
                                                       float* __restrict__ out) {
  const int idx = blockIdx.x * 256 + threadIdx.x;   // over BATCH*DIM = 16384
  float s = 0.f;
  #pragma unroll
  for (int p = 0; p < NPART; ++p) s += partials[(long)p * BATCH * DIM + idx];
  out[idx] = s;
}

// --------------------------------------------------------------------------
extern "C" void kernel_launch(void* const* d_in, const int* in_sizes, int n_in,
                              void* d_out, int out_size, void* d_ws, size_t ws_size,
                              hipStream_t stream) {
  const float* X  = (const float*)d_in[0];   // [B,T,D]
  const float* Ww = (const float*)d_in[1];   // [D,U]
  const float* Wb = (const float*)d_in[2];   // [U]
  const float* Vw = (const float*)d_in[3];   // [U,1]
  const float* Vb = (const float*)d_in[4];   // [1]
  float* out = (float*)d_out;                // [B,D]

  char* ws = (char*)d_ws;
  __bf16* Wsw   = (__bf16*)ws;                               // 128 KB
  float* logits = (float*)(ws + 131072);                     // 1 MB
  float* attn   = (float*)(ws + 131072 + 1048576);           // 1 MB
  float* parts  = (float*)(ws + 131072 + 2 * 1048576);       // 2 MB
  (void)in_sizes; (void)n_in; (void)out_size; (void)ws_size;

  prep_w_bf16    <<<DIM * DIM / 256, 256, 0, stream>>>(Ww, Wsw);
  score_logits   <<<BT / ROWS_PER_WG, 256, 0, stream>>>(X, Wsw, Wb, Vw, Vb, logits);
  softmax_t      <<<BATCH, 256, 0, stream>>>(logits, attn);
  weighted_sum   <<<BATCH * TCHUNK, 256, 0, stream>>>(X, attn, parts);
  reduce_partials<<<BATCH * DIM / 256, 256, 0, stream>>>(parts, out);
}
